// MovingAverageGatedAttention_51436528337426
// MI455X (gfx1250) — matrix operand order repaired and checked by reading, not compiled, not run on hardware
//
#include <hip/hip_runtime.h>
#include <hip/hip_bf16.h>

typedef __attribute__((ext_vector_type(16))) _Float16 v16h;
typedef __attribute__((ext_vector_type(8)))  _Float16 v8h;
typedef __attribute__((ext_vector_type(8)))  float    v8f;

#define LQ 2048
#define BQ 4
#define DQ 1024
#define ZQ 128
#define HQ 2048
#define NQ 16
#define NBQ (ZQ + HQ + 2*DQ)   // 4224
#define APITCH 40              // LDS row pitch in halves (80B, 16B-aligned, conflict-padded)

__device__ __forceinline__ float sigmf(float x) { return 1.0f / (1.0f + __expf(-x)); }
__device__ __forceinline__ float siluf(float x) { return x * sigmf(x); }
__device__ __forceinline__ v16h cat8(v8h a, v8h b) {
  return __builtin_shufflevector(a, b, 0,1,2,3,4,5,6,7,8,9,10,11,12,13,14,15);
}

// ---------------- block reductions (wave32, 8 waves / 256 threads) ----------------
__device__ __forceinline__ float blockReduceSum(float v, float* sbuf) {
  #pragma unroll
  for (int off = 16; off > 0; off >>= 1) v += __shfl_down(v, off, 32);
  int wid = threadIdx.x >> 5;
  if ((threadIdx.x & 31) == 0) sbuf[wid] = v;
  __syncthreads();
  if (threadIdx.x < 8) {
    v = sbuf[threadIdx.x];
    #pragma unroll
    for (int off = 4; off > 0; off >>= 1) v += __shfl_down(v, off, 32);
    if (threadIdx.x == 0) sbuf[0] = v;
  }
  __syncthreads();
  float r = sbuf[0];
  __syncthreads();
  return r;
}

__device__ __forceinline__ float blockReduceMax(float v, float* sbuf) {
  #pragma unroll
  for (int off = 16; off > 0; off >>= 1) v = fmaxf(v, __shfl_down(v, off, 32));
  int wid = threadIdx.x >> 5;
  if ((threadIdx.x & 31) == 0) sbuf[wid] = v;
  __syncthreads();
  if (threadIdx.x < 8) {
    v = sbuf[threadIdx.x];
    #pragma unroll
    for (int off = 4; off > 0; off >>= 1) v = fmaxf(v, __shfl_down(v, off, 32));
    if (threadIdx.x == 0) sbuf[0] = v;
  }
  __syncthreads();
  float r = sbuf[0];
  __syncthreads();
  return r;
}

// ---------------- LayerNorm: x(L,B,D) -> xn(B,L,D) fp32 + f16 ----------------
__global__ void ln_kernel(const float* __restrict__ x, const float* __restrict__ g,
                          const float* __restrict__ be, float* __restrict__ xnf,
                          _Float16* __restrict__ xn16) {
  __shared__ float sbuf[8];
  int m = blockIdx.x;            // b*L + l
  int l = m & (LQ - 1);
  int b = m >> 11;
  const float* row = x + ((size_t)l * BQ + b) * DQ;
  float v[4];
  float s = 0.f, s2 = 0.f;
  #pragma unroll
  for (int i = 0; i < 4; i++) {
    v[i] = row[threadIdx.x + i * 256];
    s += v[i]; s2 += v[i] * v[i];
  }
  s  = blockReduceSum(s,  sbuf);
  s2 = blockReduceSum(s2, sbuf);
  float mu  = s * (1.0f / DQ);
  float var = s2 * (1.0f / DQ) - mu * mu;
  float inv = rsqrtf(var + 1e-5f);
  size_t ob = ((size_t)b * LQ + l) * DQ;
  #pragma unroll
  for (int i = 0; i < 4; i++) {
    int d = threadIdx.x + i * 256;
    float xv = (v[i] - mu) * inv * g[d] + be[d];
    xnf[ob + d]  = xv;
    xn16[ob + d] = (_Float16)xv;
  }
}

// ---------------- EMA as 16-state linear recurrence -> mx16(B,L,D) ----------------
__global__ void ema_kernel(const float* __restrict__ xnf,
                           const float* __restrict__ delta, const float* __restrict__ alpha,
                           const float* __restrict__ beta,  const float* __restrict__ gamma,
                           const float* __restrict__ omega, _Float16* __restrict__ mx16) {
  int d = blockIdx.x * 256 + threadIdx.x;
  int b = blockIdx.y;
  float q[NQ], c[NQ], S[NQ];
  const float scale = 0.25f;   // sqrt(1/16)
  #pragma unroll
  for (int n = 0; n < NQ; n++) {
    float p = sigmf(delta[d * NQ + n]);
    float a = sigmf(alpha[d * NQ + n]);
    q[n] = 1.0f - p * a;
    c[n] = p * beta[d * NQ + n] * gamma[d * NQ + n] * scale;
    S[n] = 0.0f;
  }
  float om = omega[d];
  const float* xp = xnf + (size_t)b * LQ * DQ + d;
  _Float16*    mp = mx16 + (size_t)b * LQ * DQ + d;
  for (int l = 0; l < LQ; l++) {
    float xv = xp[(size_t)l * DQ];
    float y = 0.0f;
    #pragma unroll
    for (int n = 0; n < NQ; n++) {
      S[n] = fmaf(S[n], q[n], xv);
      y    = fmaf(c[n], S[n], y);
    }
    float o = y + xv * om;
    mp[(size_t)l * DQ] = (_Float16)(o * sigmf(o));   // silu
  }
}

// ---------------- transpose + f32->f16: src(R,C) -> dst(C,R) ----------------
__global__ void transpose_f32_f16(const float* __restrict__ src, _Float16* __restrict__ dst,
                                  int R, int C) {
  __shared__ float tile[16][17];
  int r = blockIdx.y * 16 + threadIdx.y;
  int c = blockIdx.x * 16 + threadIdx.x;
  tile[threadIdx.y][threadIdx.x] = src[(size_t)r * C + c];
  __syncthreads();
  int rr = blockIdx.y * 16 + threadIdx.x;
  int cc = blockIdx.x * 16 + threadIdx.y;
  dst[(size_t)cc * R + rr] = (_Float16)tile[threadIdx.x][threadIdx.y];
}

// ---------------- softmax over rows of qk(B,L,L) -> attn16 ----------------
__global__ void softmax_kernel(const float* __restrict__ qk, _Float16* __restrict__ attn) {
  __shared__ float sbuf[8];
  int row = blockIdx.x;          // b*L + l
  const float* rp = qk + (size_t)row * LQ;
  _Float16*    op = attn + (size_t)row * LQ;
  float v[8];
  float mx = -3.0e38f;
  #pragma unroll
  for (int i = 0; i < 8; i++) { v[i] = rp[threadIdx.x + i * 256]; mx = fmaxf(mx, v[i]); }
  mx = blockReduceMax(mx, sbuf);
  float s = 0.f;
  #pragma unroll
  for (int i = 0; i < 8; i++) { v[i] = __expf(v[i] - mx); s += v[i]; }
  s = blockReduceSum(s, sbuf);
  float inv = 1.0f / s;
  #pragma unroll
  for (int i = 0; i < 8; i++) op[threadIdx.x + i * 256] = (_Float16)(v[i] * inv);
}

// ---------------- LDS-staged, double-buffered WMMA GEMM with fused epilogues ----------
// A: (rows x K) f16 row-major.  Bt: (N x K) f16 (B pre-transposed, K contiguous).
// Block tile 128(M) x 128(N); 8 waves, each wave a 32x64 tile (8 wmma / K-step).
struct GP {
  const _Float16* A; const _Float16* Bt; const float* bias;
  int M, N, K;
  size_t aBatch, btBatch;
  float* f0; float* f1;
  _Float16* h0; _Float16* h1; _Float16* h2;
  const float* c0; const float* c1; const float* c2;
};

template<int EPI>
__global__ void __launch_bounds__(256) gemm_wmma(GP p) {
  __shared__ _Float16 Abuf[2][128 * APITCH];
  __shared__ _Float16 Bbuf[2][128 * APITCH];
  const int t    = threadIdx.x;
  const int wave = t >> 5;
  const int lane = t & 31;
  const bool hi  = lane >= 16;
  const int lcol = lane & 15;
  const int mw = (wave >> 1) * 32;     // 0,32,64,96
  const int nw = (wave & 1) * 64;      // 0,64
  const int blockM = blockIdx.y * 128;
  const int blockN = blockIdx.x * 128;
  const int z = blockIdx.z;
  const int K = p.K;
  const _Float16* A  = p.A  + (size_t)z * p.aBatch;
  const _Float16* Bt = p.Bt + (size_t)z * p.btBatch;

  // cooperative staging: each tile = 128 rows x 32 halves = 512 chunks of 16B.
  // thread t handles rows r0 (=t/4) and r0+64, at k-offset ko = (t%4)*8.
  const int r0 = t >> 2;
  const int r1 = r0 + 64;
  const int ko = (t & 3) * 8;
  const _Float16* gA0 = A  + (size_t)(blockM + r0) * K + ko;
  const _Float16* gA1 = A  + (size_t)(blockM + r1) * K + ko;
  const _Float16* gB0 = Bt + (size_t)(blockN + r0) * K + ko;
  const _Float16* gB1 = Bt + (size_t)(blockN + r1) * K + ko;
  const int sa0 = r0 * APITCH + ko;
  const int sa1 = r1 * APITCH + ko;

  // prologue: stage K-step 0 into buffer 0
  v8h a0 = *(const v8h*)(gA0);
  v8h a1 = *(const v8h*)(gA1);
  v8h b0 = *(const v8h*)(gB0);
  v8h b1 = *(const v8h*)(gB1);
  *(v8h*)&Abuf[0][sa0] = a0;
  *(v8h*)&Abuf[0][sa1] = a1;
  *(v8h*)&Bbuf[0][sa0] = b0;
  *(v8h*)&Bbuf[0][sa1] = b1;
  __syncthreads();

  v8f acc[2][4] = {};
  const int aoff = hi ? 8 : 0;     // A 16x32 f16 layout
  const int boff = hi ? 8 : 0;     // B 32x16 f16 layout
  int buf = 0;
  for (int kb = 0; kb < K; kb += 32) {
    const bool more = (kb + 32) < K;
    if (more) {                     // issue next tile's global loads early
      a0 = *(const v8h*)(gA0 + kb + 32);
      a1 = *(const v8h*)(gA1 + kb + 32);
      b0 = *(const v8h*)(gB0 + kb + 32);
      b1 = *(const v8h*)(gB1 + kb + 32);
      if (kb + 64 < K) {
        __builtin_prefetch(gA0 + kb + 64, 0, 2);   // global_prefetch_b8
        __builtin_prefetch(gB0 + kb + 64, 0, 2);
      }
    }
    const _Float16* sA = Abuf[buf];
    const _Float16* sB = Bbuf[buf];
    // load ALL fragments first so ds waits can be partial and WMMAs stream
    v16h af[2];
    #pragma unroll
    for (int mt = 0; mt < 2; mt++) {
      const _Float16* ap = sA + (mw + mt * 16 + lcol) * APITCH + aoff;
      af[mt] = cat8(*(const v8h*)(ap), *(const v8h*)(ap + 16));
    }
    v16h bf[4];
    #pragma unroll
    for (int s = 0; s < 4; s++) {
      const _Float16* bp = sB + (nw + s * 16 + lcol) * APITCH + boff;
      bf[s] = cat8(*(const v8h*)(bp), *(const v8h*)(bp + 16));
    }
    #pragma unroll
    for (int s = 0; s < 4; s++) {
      acc[0][s] = __builtin_amdgcn_wmma_f32_16x16x32_f16(
          false, af[0], false, bf[s], (short)0, acc[0][s], false, false);
      acc[1][s] = __builtin_amdgcn_wmma_f32_16x16x32_f16(
          false, af[1], false, bf[s], (short)0, acc[1][s], false, false);
    }
    if (more) {                     // stage next tile into the other buffer
      buf ^= 1;
      *(v8h*)&Abuf[buf][sa0] = a0;
      *(v8h*)&Abuf[buf][sa1] = a1;
      *(v8h*)&Bbuf[buf][sa0] = b0;
      *(v8h*)&Bbuf[buf][sa1] = b1;
    }
    __syncthreads();
  }

  // D layout: lane holds column N=(lane&15); VGPR i -> row M=i (lanes<16) / M=8+i.
  const int roff = hi ? 8 : 0;
  #pragma unroll
  for (int mt = 0; mt < 2; mt++) {
    const int m0 = blockM + mw + mt * 16;
    int b, l0;
    if constexpr (EPI == 2 || EPI == 3) { b = z; l0 = m0; }
    else { b = m0 >> 11; l0 = m0 & (LQ - 1); }

    #pragma unroll
    for (int s = 0; s < 4; s++) {
      const int n = blockN + nw + s * 16 + lcol;
      if constexpr (EPI == 0) {
        // v = silu(xn@v_w + v_b) -> f16, stored transposed per batch: vT(B,H,L)
        float bn = p.bias[n];
        v8h pk;
        #pragma unroll
        for (int i = 0; i < 8; i++) pk[i] = (_Float16)siluf(acc[mt][s][i] + bn);
        *(v8h*)(p.h0 + ((size_t)b * HQ + n) * LQ + l0 + roff) = pk;
      } else if constexpr (EPI == 1) {
        // base = mx@mx_w + mx_b, 4-way column split
        float bn = p.bias[n];
        if (n < DQ) {                                     // u = sigmoid
          float* up = p.f0 + ((size_t)b * LQ + l0 + roff) * DQ + n;
          #pragma unroll
          for (int i = 0; i < 8; i++) up[(size_t)i * DQ] = sigmf(acc[mt][s][i] + bn);
        } else if (n < DQ + ZQ) {                         // z -> q,k (gamma/beta + scaling)
          int zc = n - DQ;
          float g0 = p.c0[zc], bb0 = p.c1[zc];
          float g1 = p.c0[ZQ + zc], bb1 = p.c1[ZQ + zc];
          const float rs = 0.08838834764831845f;          // Z^-0.5
          _Float16* qp = p.h0 + ((size_t)b * LQ + l0 + roff) * ZQ + zc;
          _Float16* kp = p.h1 + ((size_t)b * LQ + l0 + roff) * ZQ + zc;
          #pragma unroll
          for (int i = 0; i < 8; i++) {
            float zz = siluf(acc[mt][s][i] + bn);
            qp[(size_t)i * ZQ] = (_Float16)((zz * g0 + bb0) * rs);
            kp[(size_t)i * ZQ] = (_Float16)(zz * g1 + bb1);
          }
        } else if (n < DQ + ZQ + HQ) {                    // r = silu
          int rc = n - DQ - ZQ;
          _Float16* rp = p.h2 + ((size_t)b * LQ + l0 + roff) * HQ + rc;
          #pragma unroll
          for (int i = 0; i < 8; i++) rp[(size_t)i * HQ] = (_Float16)siluf(acc[mt][s][i] + bn);
        } else {                                          // hx passthrough fp32
          int hc = n - DQ - ZQ - HQ;
          float* hp = p.f1 + ((size_t)b * LQ + l0 + roff) * DQ + hc;
          #pragma unroll
          for (int i = 0; i < 8; i++) hp[(size_t)i * DQ] = acc[mt][s][i] + bn;
        }
      } else if constexpr (EPI == 2) {
        // qk = q@k^T + rel_b[n - l + L - 1]
        float* op = p.f0 + ((size_t)b * LQ + l0 + roff) * LQ + n;
        #pragma unroll
        for (int i = 0; i < 8; i++) {
          int lrow = l0 + roff + i;
          op[(size_t)i * LQ] = acc[mt][s][i] + p.c0[n - lrow + LQ - 1];
        }
      } else if constexpr (EPI == 3) {
        // g = (attn@v) * r  -> f16 (B,L,H)
        size_t rb = ((size_t)b * LQ + l0 + roff) * HQ + n;
        #pragma unroll
        for (int i = 0; i < 8; i++) {
          float rv = (float)p.h1[rb + (size_t)i * HQ];
          p.h0[rb + (size_t)i * HQ] = (_Float16)(acc[mt][s][i] * rv);
        }
      } else {
        // EPI 4: h = silu(hx + g@h_w + h_b); out = res + u*(h - res), out(L,B,D)
        float bn = p.bias[n];
        size_t ub = ((size_t)b * LQ + l0 + roff) * DQ + n;
        #pragma unroll
        for (int i = 0; i < 8; i++) {
          int lrow = l0 + roff + i;
          float hval = siluf(acc[mt][s][i] + bn + p.c0[ub + (size_t)i * DQ]);
          float uv   = p.c1[ub + (size_t)i * DQ];
          size_t xi  = ((size_t)lrow * BQ + b) * DQ + n;
          float res  = p.c2[xi];
          p.f0[xi] = res + uv * (hval - res);
        }
      }
    }
  }
}

// ---------------- launcher ----------------
extern "C" void kernel_launch(void* const* d_in, const int* in_sizes, int n_in,
                              void* d_out, int out_size, void* d_ws, size_t ws_size,
                              hipStream_t stream) {
  (void)in_sizes; (void)n_in; (void)out_size; (void)ws_size;
  const float* x      = (const float*)d_in[0];
  const float* ln_g   = (const float*)d_in[1];
  const float* ln_b   = (const float*)d_in[2];
  const float* v_w    = (const float*)d_in[3];
  const float* v_b    = (const float*)d_in[4];
  const float* mx_w   = (const float*)d_in[5];
  const float* mx_b   = (const float*)d_in[6];
  const float* h_w    = (const float*)d_in[7];
  const float* h_b    = (const float*)d_in[8];
  const float* qk_g   = (const float*)d_in[9];
  const float* qk_b   = (const float*)d_in[10];
  const float* rel_b  = (const float*)d_in[11];
  const float* e_del  = (const float*)d_in[12];
  const float* e_alp  = (const float*)d_in[13];
  const float* e_bet  = (const float*)d_in[14];
  const float* e_gam  = (const float*)d_in[15];
  const float* e_ome  = (const float*)d_in[16];
  float* out = (float*)d_out;

  char* w = (char*)d_ws;
  auto take = [&](size_t bytes) -> char* {
    char* r = w;
    w += (bytes + 255) & ~(size_t)255;
    return r;
  };
  const size_t MX = (size_t)BQ * LQ;   // 8192 rows
  float*    xnf  = (float*)   take(MX * DQ * 4);
  _Float16* xn16 = (_Float16*)take(MX * DQ * 2);
  _Float16* mx16 = (_Float16*)take(MX * DQ * 2);
  _Float16* vwT  = (_Float16*)take((size_t)HQ * DQ * 2);
  _Float16* mxwT = (_Float16*)take((size_t)NBQ * DQ * 2);
  _Float16* hwT  = (_Float16*)take((size_t)DQ * HQ * 2);
  _Float16* vT   = (_Float16*)take((size_t)BQ * HQ * LQ * 2);
  float*    u    = (float*)   take(MX * DQ * 4);
  float*    hx   = (float*)   take(MX * DQ * 4);
  _Float16* q16  = (_Float16*)take(MX * ZQ * 2);
  _Float16* k16  = (_Float16*)take(MX * ZQ * 2);
  _Float16* r16  = (_Float16*)take(MX * HQ * 2);
  float*    qk   = (float*)   take((size_t)BQ * LQ * LQ * 4);
  _Float16* attn = (_Float16*)take((size_t)BQ * LQ * LQ * 2);
  _Float16* g16  = (_Float16*)take(MX * HQ * 2);

  dim3 tb(16, 16);
  transpose_f32_f16<<<dim3(HQ / 16,  DQ / 16), tb, 0, stream>>>(v_w,  vwT,  DQ, HQ);
  transpose_f32_f16<<<dim3(NBQ / 16, DQ / 16), tb, 0, stream>>>(mx_w, mxwT, DQ, NBQ);
  transpose_f32_f16<<<dim3(DQ / 16,  HQ / 16), tb, 0, stream>>>(h_w,  hwT,  HQ, DQ);

  ln_kernel<<<(int)MX, 256, 0, stream>>>(x, ln_g, ln_b, xnf, xn16);
  ema_kernel<<<dim3(DQ / 256, BQ), 256, 0, stream>>>(xnf, e_del, e_alp, e_bet, e_gam, e_ome, mx16);

  { GP p = {}; p.A = xn16; p.Bt = vwT; p.bias = v_b;
    p.M = (int)MX; p.N = HQ; p.K = DQ; p.h0 = vT;
    gemm_wmma<0><<<dim3(HQ / 128, (int)MX / 128, 1), 256, 0, stream>>>(p); }

  { GP p = {}; p.A = mx16; p.Bt = mxwT; p.bias = mx_b;
    p.M = (int)MX; p.N = NBQ; p.K = DQ;
    p.f0 = u; p.f1 = hx; p.h0 = q16; p.h1 = k16; p.h2 = r16; p.c0 = qk_g; p.c1 = qk_b;
    gemm_wmma<1><<<dim3(NBQ / 128, (int)MX / 128, 1), 256, 0, stream>>>(p); }

  { GP p = {}; p.A = q16; p.Bt = k16;
    p.M = LQ; p.N = LQ; p.K = ZQ;
    p.aBatch = (size_t)LQ * ZQ; p.btBatch = (size_t)LQ * ZQ;
    p.f0 = qk; p.c0 = rel_b;
    gemm_wmma<2><<<dim3(LQ / 128, LQ / 128, BQ), 256, 0, stream>>>(p); }

  softmax_kernel<<<(int)MX, 256, 0, stream>>>(qk, attn);

  { GP p = {}; p.A = attn; p.Bt = vT;
    p.M = LQ; p.N = HQ; p.K = LQ;
    p.aBatch = (size_t)LQ * LQ; p.btBatch = (size_t)HQ * LQ;
    p.h0 = g16; p.h1 = r16;
    gemm_wmma<3><<<dim3(HQ / 128, LQ / 128, BQ), 256, 0, stream>>>(p); }

  { GP p = {}; p.A = g16; p.Bt = hwT; p.bias = h_b;
    p.M = (int)MX; p.N = DQ; p.K = HQ;
    p.c0 = hx; p.c1 = u; p.c2 = x; p.f0 = out;
    gemm_wmma<4><<<dim3(DQ / 128, (int)MX / 128, 1), 256, 0, stream>>>(p); }
}
